// GATPolicy_10574209483592
// MI455X (gfx1250) — matrix operand order repaired and checked by reading, not compiled, not run on hardware
//
#include <hip/hip_runtime.h>
#include <hip/hip_bf16.h>

#define NN    50000
#define EE    800000
#define EF    850000      /* E + N self-loops */
#define FIN   64
#define ED    16
#define HH    4
#define CC    32
#define HC    128
#define BB    64
#define AA    8
#define SLOPE 0.2f
#define TPB   256
#define GEMM_BLOCKS 512

typedef __attribute__((ext_vector_type(16))) _Float16 v16h;
typedef __attribute__((ext_vector_type(8)))  _Float16 v8h;
typedef __attribute__((ext_vector_type(2)))  _Float16 v2h;
typedef __attribute__((ext_vector_type(8)))  float    v8f;

// ---------- monotone float<->uint encoding for order-independent atomic max ----------
__device__ __forceinline__ unsigned fenc(float f) {
    unsigned u = __float_as_uint(f);
    return (u & 0x80000000u) ? ~u : (u | 0x80000000u);
}
__device__ __forceinline__ float fdec(unsigned u) {
    unsigned v = (u & 0x80000000u) ? (u & 0x7FFFFFFFu) : ~u;
    return __uint_as_float(v);
}
#define ENC_NEG_INF 0x007FFFFFu   /* fenc(-inf) */

// ---------- fills ----------
__global__ void k_fill_f32(float* p, float v, int n) {
    int i = blockIdx.x * blockDim.x + threadIdx.x;
    if (i < n) p[i] = v;
}
__global__ void k_fill_u32(unsigned* p, unsigned v, int n) {
    int i = blockIdx.x * blockDim.x + threadIdx.x;
    if (i < n) p[i] = v;
}

// ---------- self-loop mean edge_attr ----------
__global__ void k_edge_mean(const int* __restrict__ dst, const float* __restrict__ eattr,
                            float* __restrict__ cnt, float* __restrict__ msum) {
    int e = blockIdx.x * blockDim.x + threadIdx.x;
    if (e >= EE) return;
    int d = dst[e];
    atomicAdd(&cnt[d], 1.0f);
    const float* row = eattr + (size_t)e * ED;
    float* out = msum + (size_t)d * ED;
    #pragma unroll
    for (int j = 0; j < ED; ++j) atomicAdd(&out[j], row[j]);
}
__global__ void k_mean_div(float* __restrict__ msum, const float* __restrict__ cnt) {
    int i = blockIdx.x * blockDim.x + threadIdx.x;
    if (i >= NN * ED) return;
    float c = cnt[i >> 4];
    msum[i] /= fmaxf(c, 1.0f);
}

// ---------- WMMA GEMM: out[NN,HC] = in[NN,Kc] @ W[Kc,HC] ----------
// 256 threads = 8 waves; wave w owns column tile w and holds its B fragments in
// registers for the whole kernel; block grid-strides over 16-row tiles, staging
// each activation tile through LDS as f16 (ds_store_b32 in, ds_load_b128 out).
template <int Kc>
__global__ __launch_bounds__(256) void k_gemm_wmma(const float* __restrict__ in,
                                                   const float* __restrict__ W,
                                                   float* __restrict__ out) {
    constexpr int KST = Kc + 8;                       // padded LDS row stride (halves)
    __shared__ __align__(16) _Float16 As[16 * KST];
    const int tid  = threadIdx.x;
    const int wave = tid >> 5;                        // 0..7 -> column tile
    const int lane = tid & 31;
    const int g    = lane >> 4;                       // lane group
    const int lm   = lane & 15;
    const int coln = wave * 16 + lm;                  // B/D column owned by this lane

    // B fragments (32x16 f16 per K-step): lanes 0-15 hold K=0..15, lanes 16-31 K=16..31
    v16h bfrag[Kc / 32];
    #pragma unroll
    for (int s = 0; s < Kc / 32; ++s) {
        #pragma unroll
        for (int v = 0; v < 8; ++v) {
            int kb = s * 32 + g * 16 + 2 * v;
            bfrag[s][2 * v]     = (_Float16)W[(size_t)kb * HC + coln];
            bfrag[s][2 * v + 1] = (_Float16)W[(size_t)(kb + 1) * HC + coln];
        }
    }

    for (int tile = blockIdx.x; tile < NN / 16; tile += gridDim.x) {
        __syncthreads();                              // protect LDS from previous iter
        // cooperative coalesced stage: 16 rows x Kc f32 -> packed f16 in LDS
        for (int p = tid; p < 16 * (Kc / 2); p += 256) {
            int row = p / (Kc / 2);
            int kp  = p % (Kc / 2);
            const float* sp = in + (size_t)(tile * 16 + row) * Kc + 2 * kp;
            v2h pk;
            pk[0] = (_Float16)sp[0];
            pk[1] = (_Float16)sp[1];
            *(v2h*)&As[row * KST + 2 * kp] = pk;
        }
        __syncthreads();

        v8f c = {};
        #pragma unroll
        for (int s = 0; s < Kc / 32; ++s) {
            // A 16x32 f16 fragment: lane holds one contiguous 16-half run
            const _Float16* ab = &As[lm * KST + s * 32 + g * 16];
            v8h lo = *(const v8h*)(ab);               // K = s*32 + g*16 + 0..7
            v8h hi = *(const v8h*)(ab + 8);           // K = s*32 + g*16 + 8..15
            v16h a = __builtin_shufflevector(lo, hi, 0, 1, 2, 3, 4, 5, 6, 7,
                                                     8, 9, 10, 11, 12, 13, 14, 15);
            c = __builtin_amdgcn_wmma_f32_16x16x32_f16(false, a, false, bfrag[s],
                                                       (short)0, c, false, false);
        }
        #pragma unroll
        for (int r = 0; r < 8; ++r) {                 // D: VGPR r -> M = r + 8*g
            out[(size_t)(tile * 16 + r + 8 * g) * HC + coln] = c[r];
        }
    }
}

// ---------- per-node attention scores sc_s, sc_d ----------
__global__ void k_scores_node(const float* __restrict__ xw, const float* __restrict__ as_,
                              const float* __restrict__ ad_, float* __restrict__ scs,
                              float* __restrict__ scd) {
    int i = blockIdx.x * blockDim.x + threadIdx.x;  // n*H + h
    if (i >= NN * HH) return;
    int n = i >> 2, h = i & 3;
    const float* xr = xw + (size_t)n * HC + h * CC;
    const float* av = as_ + h * CC;
    const float* bv = ad_ + h * CC;
    float s0 = 0.f, s1 = 0.f;
    #pragma unroll
    for (int cidx = 0; cidx < CC; ++cidx) { s0 += xr[cidx] * av[cidx]; s1 += xr[cidx] * bv[cidx]; }
    scs[i] = s0; scd[i] = s1;
}

// ---------- tiny M[d][h] = sum_c We[d, h*32+c] * ae[h,c] ----------
__global__ void k_eM(const float* __restrict__ We, const float* __restrict__ ae,
                     float* __restrict__ eM) {
    int i = threadIdx.x;                 // 64 threads: d*4 + h
    if (i >= ED * HH) return;
    int d = i >> 2, h = i & 3;
    float s = 0.f;
    #pragma unroll
    for (int cidx = 0; cidx < CC; ++cidx) s += We[(size_t)d * HC + h * CC + cidx] * ae[h * CC + cidx];
    eM[i] = s;
}

// ---------- sc_e[e,h] = attr(e,:) @ M[:,h] (self-loops use mean_attr) ----------
__global__ void k_sce(const float* __restrict__ eattr, const float* __restrict__ mattr,
                      const float* __restrict__ eM, float* __restrict__ sce) {
    int i = blockIdx.x * blockDim.x + threadIdx.x;   // e*H + h
    if (i >= EF * HH) return;
    int e = i >> 2, h = i & 3;
    const float* row = (e < EE) ? (eattr + (size_t)e * ED) : (mattr + (size_t)(e - EE) * ED);
    float s = 0.f;
    #pragma unroll
    for (int d = 0; d < ED; ++d) s += row[d] * eM[d * HH + h];
    sce[i] = s;
}

// ---------- raw alpha (leaky relu) + segment max ----------
__global__ void k_alpha_max(const int* __restrict__ src, const int* __restrict__ dst,
                            const float* __restrict__ scs, const float* __restrict__ scd,
                            float* __restrict__ alpha, unsigned* __restrict__ menc) {
    int i = blockIdx.x * blockDim.x + threadIdx.x;   // e*H + h
    if (i >= EF * HH) return;
    int e = i >> 2, h = i & 3;
    int sV = (e < EE) ? src[e] : (e - EE);
    int dV = (e < EE) ? dst[e] : (e - EE);
    float a = scs[sV * HH + h] + scd[dV * HH + h] + alpha[i];  // alpha currently holds sc_e
    a = (a > 0.f) ? a : SLOPE * a;
    alpha[i] = a;
    atomicMax(&menc[dV * HH + h], fenc(a));
}

// ---------- exp(alpha - m) + segment sum ----------
__global__ void k_alpha_exp(const int* __restrict__ src, const int* __restrict__ dst,
                            const unsigned* __restrict__ menc, float* __restrict__ alpha,
                            float* __restrict__ ssum) {
    int i = blockIdx.x * blockDim.x + threadIdx.x;
    if (i >= EF * HH) return;
    int e = i >> 2, h = i & 3;
    int dV = (e < EE) ? dst[e] : (e - EE);
    float mm = fdec(menc[dV * HH + h]);
    if (mm < -3.0e38f) mm = 0.f;                     // non-finite guard (matches reference)
    float a = __expf(alpha[i] - mm);
    alpha[i] = a;
    atomicAdd(&ssum[dV * HH + h], a);
}

// ---------- message aggregation: acc[dst,t] += (alpha/s) * xw[src,t] ----------
__global__ void k_aggregate(const int* __restrict__ src, const int* __restrict__ dst,
                            const float* __restrict__ alpha, const float* __restrict__ ssum,
                            const float* __restrict__ xw, float* __restrict__ acc) {
    long long i = (long long)blockIdx.x * blockDim.x + threadIdx.x;   // e*128 + t
    if (i >= (long long)EF * HC) return;
    int e = (int)(i >> 7), t = (int)(i & 127), h = t >> 5;
    int sV = (e < EE) ? src[e] : (e - EE);
    int dV = (e < EE) ? dst[e] : (e - EE);
    float w = alpha[e * HH + h] / (ssum[dV * HH + h] + 1e-16f);
    atomicAdd(&acc[(size_t)dV * HC + t], w * xw[(size_t)sV * HC + t]);
}

// ---------- bias + relu (in place) ----------
__global__ void k_bias_relu(float* __restrict__ acc, const float* __restrict__ bias) {
    int i = blockIdx.x * blockDim.x + threadIdx.x;
    if (i >= NN * HC) return;
    acc[i] = fmaxf(acc[i] + bias[i & (HC - 1)], 0.f);
}

// ---------- global mean pool + tanh head ----------
__global__ void k_pool_cnt(const int* __restrict__ batch, float* __restrict__ gcnt) {
    int n = blockIdx.x * blockDim.x + threadIdx.x;
    if (n < NN) atomicAdd(&gcnt[batch[n]], 1.0f);
}
__global__ void k_pool_sum(const int* __restrict__ batch, const float* __restrict__ acc,
                           float* __restrict__ gsum) {
    int i = blockIdx.x * blockDim.x + threadIdx.x;   // n*128 + t
    if (i >= NN * HC) return;
    int n = i >> 7, t = i & 127;
    atomicAdd(&gsum[batch[n] * HC + t], acc[i]);
}
__global__ void k_head(const float* __restrict__ gsum, const float* __restrict__ gcnt,
                       const float* __restrict__ Wl, const float* __restrict__ bl,
                       float* __restrict__ out) {
    int i = blockIdx.x * blockDim.x + threadIdx.x;   // b*8 + a
    if (i >= BB * AA) return;
    int b = i >> 3, a = i & 7;
    float inv = 1.0f / fmaxf(gcnt[b], 1.0f);
    float s = bl[a];
    #pragma unroll
    for (int ch = 0; ch < HC; ++ch) s += gsum[b * HC + ch] * inv * Wl[ch * AA + a];
    out[i] = tanhf(s);
}

static inline int nblk(long long n) { return (int)((n + TPB - 1) / TPB); }

extern "C" void kernel_launch(void* const* d_in, const int* in_sizes, int n_in,
                              void* d_out, int out_size, void* d_ws, size_t ws_size,
                              hipStream_t stream) {
    const float* x     = (const float*)d_in[0];
    const float* eattr = (const float*)d_in[1];
    const int*   eidx  = (const int*)d_in[2];
    const int*   batch = (const int*)d_in[3];
    const int*   src   = eidx;
    const int*   dstp  = eidx + EE;
    const float* Wl    = (const float*)d_in[22];
    const float* bl    = (const float*)d_in[23];

    // workspace carve-up (256B aligned)
    char* ws = (char*)d_ws;
    size_t off = 0;
    auto carve = [&](size_t bytes) { void* p = ws + off; off = (off + bytes + 255) & ~(size_t)255; return p; };
    float*    cnt   = (float*)   carve((size_t)NN * 4);
    float*    mattr = (float*)   carve((size_t)NN * ED * 4);
    float*    xw    = (float*)   carve((size_t)NN * HC * 4);
    float*    acc   = (float*)   carve((size_t)NN * HC * 4);
    float*    scs   = (float*)   carve((size_t)NN * HH * 4);
    float*    scd   = (float*)   carve((size_t)NN * HH * 4);
    float*    eM    = (float*)   carve((size_t)ED * HH * 4);
    float*    alpha = (float*)   carve((size_t)EF * HH * 4);   // holds sc_e, then alpha
    unsigned* menc  = (unsigned*)carve((size_t)NN * HH * 4);
    float*    ssum  = (float*)   carve((size_t)NN * HH * 4);
    float*    gsum  = (float*)   carve((size_t)BB * HC * 4);
    float*    gcnt  = (float*)   carve((size_t)BB * 4);
    (void)ws_size; (void)in_sizes; (void)n_in; (void)out_size;

    // self-loop mean edge attributes
    k_fill_f32<<<nblk(NN), TPB, 0, stream>>>(cnt, 0.f, NN);
    k_fill_f32<<<nblk((long long)NN * ED), TPB, 0, stream>>>(mattr, 0.f, NN * ED);
    k_edge_mean<<<nblk(EE), TPB, 0, stream>>>(dstp, eattr, cnt, mattr);
    k_mean_div<<<nblk((long long)NN * ED), TPB, 0, stream>>>(mattr, cnt);

    for (int l = 0; l < 3; ++l) {
        const float* W   = (const float*)d_in[4 + 6 * l];
        const float* We  = (const float*)d_in[5 + 6 * l];
        const float* as_ = (const float*)d_in[6 + 6 * l];
        const float* ad_ = (const float*)d_in[7 + 6 * l];
        const float* ae  = (const float*)d_in[8 + 6 * l];
        const float* bia = (const float*)d_in[9 + 6 * l];

        if (l == 0)
            k_gemm_wmma<FIN><<<GEMM_BLOCKS, 256, 0, stream>>>(x, W, xw);
        else
            k_gemm_wmma<HC><<<GEMM_BLOCKS, 256, 0, stream>>>(acc, W, xw);
        k_scores_node<<<nblk((long long)NN * HH), TPB, 0, stream>>>(xw, as_, ad_, scs, scd);
        k_eM<<<1, 64, 0, stream>>>(We, ae, eM);
        k_sce<<<nblk((long long)EF * HH), TPB, 0, stream>>>(eattr, mattr, eM, alpha);
        k_fill_u32<<<nblk((long long)NN * HH), TPB, 0, stream>>>(menc, ENC_NEG_INF, NN * HH);
        k_fill_f32<<<nblk((long long)NN * HH), TPB, 0, stream>>>(ssum, 0.f, NN * HH);
        k_fill_f32<<<nblk((long long)NN * HC), TPB, 0, stream>>>(acc, 0.f, NN * HC); // after GEMM read
        k_alpha_max<<<nblk((long long)EF * HH), TPB, 0, stream>>>(src, dstp, scs, scd, alpha, menc);
        k_alpha_exp<<<nblk((long long)EF * HH), TPB, 0, stream>>>(src, dstp, menc, alpha, ssum);
        k_aggregate<<<nblk((long long)EF * HC), TPB, 0, stream>>>(src, dstp, alpha, ssum, xw, acc);
        k_bias_relu<<<nblk((long long)NN * HC), TPB, 0, stream>>>(acc, bia);
    }

    k_fill_f32<<<nblk(BB * HC), TPB, 0, stream>>>(gsum, 0.f, BB * HC);
    k_fill_f32<<<1, BB, 0, stream>>>(gcnt, 0.f, BB);
    k_pool_cnt<<<nblk(NN), TPB, 0, stream>>>(batch, gcnt);
    k_pool_sum<<<nblk((long long)NN * HC), TPB, 0, stream>>>(batch, acc, gsum);
    k_head<<<nblk(BB * AA), TPB, 0, stream>>>(gsum, gcnt, Wl, bl, (float*)d_out);
}
